// GATEncoder_82566451298971
// MI455X (gfx1250) — matrix operand order repaired and checked by reading, not compiled, not run on hardware
//
#include <hip/hip_runtime.h>
#include <math.h>

typedef __attribute__((ext_vector_type(16))) __bf16 v16bf;
typedef __attribute__((ext_vector_type(8)))  float  v8f;
typedef __attribute__((ext_vector_type(4)))  unsigned int u32x4;
typedef __attribute__((ext_vector_type(4)))  float  f32x4;

#define NNODES 10000
#define NEDGES 160000
#define ETOT   (NEDGES + NNODES)
#define INDIM  512
#define F1     1024       // HEADS*HID
#define H1     8
#define C1     128
#define F2     256
#define NEG_SLOPE 0.2f

// ---------- order-preserving float<->uint encoding for atomic max ----------
static __device__ __forceinline__ unsigned int enc_f32(float f) {
  unsigned int u = __float_as_uint(f);
  return (u & 0x80000000u) ? ~u : (u | 0x80000000u);
}
static __device__ __forceinline__ float dec_f32(unsigned int u) {
  return __uint_as_float((u & 0x80000000u) ? (u ^ 0x80000000u) : ~u);
}

// ---------- elementwise helpers ----------
__global__ void k_cvt_bf16(const float* __restrict__ s, __bf16* __restrict__ d, int n) {
  int i = blockIdx.x * blockDim.x + threadIdx.x;
  if (i < n) d[i] = (__bf16)s[i];
}

__global__ void k_fill_u32(unsigned int* __restrict__ p, unsigned int v, int n) {
  int i = blockIdx.x * blockDim.x + threadIdx.x;
  if (i < n) p[i] = v;
}

__global__ void k_elu_cvt(const float* __restrict__ a, const float* __restrict__ b,
                          __bf16* __restrict__ o, int n, int rowMask) {
  int i = blockIdx.x * blockDim.x + threadIdx.x;
  if (i >= n) return;
  float x = a[i] + b[i & rowMask];
  x = (x > 0.f) ? x : (expf(x) - 1.f);
  o[i] = (__bf16)x;
}

__global__ void k_add_bias(float* __restrict__ o, const float* __restrict__ b,
                           int n, int rowMask) {
  int i = blockIdx.x * blockDim.x + threadIdx.x;
  if (i < n) o[i] = o[i] + b[i & rowMask];
}

// ---------- WMMA fragment gathers (contiguous k = 16*(lane/16) + element, same for A and B) ----------
// A (16x32, MxK): lanes 0-15 m=0..15; VGPR v -> K=16g+2v(,+1)
static __device__ __forceinline__ v16bf frag_a(const __bf16* __restrict__ p, int g) {
  v16bf f;
#pragma unroll
  for (int v = 0; v < 4; ++v) {
    int k0 = 16 * g + 2 * v;
    f[2 * v] = p[k0]; f[2 * v + 1] = p[k0 + 1];
  }
#pragma unroll
  for (int v = 4; v < 8; ++v) {
    int k0 = 16 * g + 2 * v;
    f[2 * v] = p[k0]; f[2 * v + 1] = p[k0 + 1];
  }
  return f;
}
// B (32x16, KxN) with Bs stored [n][k]: lanes 0-15 K=0..15, lanes 16-31 K=16..31
static __device__ __forceinline__ v16bf frag_b(const __bf16* __restrict__ p, int g) {
  v16bf f;
#pragma unroll
  for (int v = 0; v < 8; ++v) {
    int k0 = 16 * g + 2 * v;
    f[2 * v] = p[k0]; f[2 * v + 1] = p[k0 + 1];
  }
  return f;
}

// ---------- bf16 WMMA GEMM: C[M,N] = A[M,K] * B[K,N] ----------
// 256 threads = 8 waves (2x4). Block tile 64(M) x 128(N); wave tile 32x32 (2x2 WMMA).
__global__ __launch_bounds__(256)
void k_gemm_bf16(const __bf16* __restrict__ A, const __bf16* __restrict__ B,
                 float* __restrict__ C, int M, int N, int K) {
  __shared__ __bf16 As[64][40];    // [m][k]
  __shared__ __bf16 Bs[128][40];   // [n][k] (transposed on store)

  const int tid  = threadIdx.x;
  const int lane = tid & 31;
  const int wav  = tid >> 5;
  const int wm   = wav >> 2;          // 0..1 -> M offset wm*32
  const int wn   = wav & 3;           // 0..3 -> N offset wn*32
  const int g    = lane >> 4;
  const int r16  = lane & 15;

  const int Mbase = blockIdx.y * 64;
  const int Nbase = blockIdx.x * 128;

  const int arow = tid >> 2;          // 0..63
  const int acol = (tid & 3) * 8;     // 0,8,16,24
  const int kb   = tid >> 4;          // 0..15 (B k rows; also kb+16)
  const int bcol = (tid & 15) * 8;    // 0..120

  const int gmA = Mbase + arow;
  const __bf16* aSrc = A + (size_t)gmA * K + acol;
  const __bf16* bSrc = B + (size_t)kb * N + Nbase + bcol;
  const size_t bRowStride = (size_t)N;

  v8f acc00 = {}, acc01 = {}, acc10 = {}, acc11 = {};

  for (int kt = 0; kt < K; kt += 32) {
    // ---- stage A (64x32), zero-pad M edge ----
    u32x4 av = {};
    if (gmA < M) av = *(const u32x4*)(aSrc + kt);
    *(u32x4*)(&As[arow][acol]) = av;
    // ---- stage B (32x128) transposed into Bs[n][k] ----
    union { u32x4 u; __bf16 h[8]; } b0, b1;
    b0.u = *(const u32x4*)(bSrc + (size_t)kt * bRowStride);
    b1.u = *(const u32x4*)(bSrc + (size_t)(kt + 16) * bRowStride);
#pragma unroll
    for (int j = 0; j < 8; ++j) {
      Bs[bcol + j][kb]      = b0.h[j];
      Bs[bcol + j][kb + 16] = b1.h[j];
    }
    // prefetch next K tile into caches while this stage computes
    if (kt + 32 < K) {
      if (gmA < M) __builtin_prefetch(aSrc + kt + 32, 0, 3);
      __builtin_prefetch(bSrc + (size_t)(kt + 32) * bRowStride, 0, 3);
      __builtin_prefetch(bSrc + (size_t)(kt + 48) * bRowStride, 0, 3);
    }
    __syncthreads();

    v16bf a0 = frag_a(&As[wm * 32 + r16][0], g);
    v16bf a1 = frag_a(&As[wm * 32 + 16 + r16][0], g);
    v16bf bq0 = frag_b(&Bs[wn * 32 + r16][0], g);
    v16bf bq1 = frag_b(&Bs[wn * 32 + 16 + r16][0], g);

    acc00 = __builtin_amdgcn_wmma_f32_16x16x32_bf16(false, a0, false, bq0, (short)0, acc00, false, false);
    acc01 = __builtin_amdgcn_wmma_f32_16x16x32_bf16(false, a0, false, bq1, (short)0, acc01, false, false);
    acc10 = __builtin_amdgcn_wmma_f32_16x16x32_bf16(false, a1, false, bq0, (short)0, acc10, false, false);
    acc11 = __builtin_amdgcn_wmma_f32_16x16x32_bf16(false, a1, false, bq1, (short)0, acc11, false, false);
    __syncthreads();
  }

  // C/D layout: n = lane%16, m = r + 8*(lane/16)
#pragma unroll
  for (int r = 0; r < 8; ++r) {
    int m0 = Mbase + wm * 32 + g * 8 + r;
    if (m0 < M) {
      float* crow = C + (size_t)m0 * N + Nbase + wn * 32 + r16;
      crow[0]  = acc00[r];
      crow[16] = acc01[r];
    }
    int m1 = m0 + 16;
    if (m1 < M) {
      float* crow = C + (size_t)m1 * N + Nbase + wn * 32 + r16;
      crow[0]  = acc10[r];
      crow[16] = acc11[r];
    }
  }
}

// ---------- attention scalar scores ----------
__global__ void k_attn_scores(const float* __restrict__ h,
                              const float* __restrict__ a_s,
                              const float* __restrict__ a_d,
                              float* __restrict__ os, float* __restrict__ od,
                              int N, int H, int C) {
  int i = blockIdx.x * blockDim.x + threadIdx.x;
  if (i >= N * H) return;
  int n  = i / H;
  int hh = i % H;
  const float* hp = h + (size_t)n * H * C + (size_t)hh * C;
  const float* as = a_s + (size_t)hh * C;
  const float* ad = a_d + (size_t)hh * C;
  float ss = 0.f, sd = 0.f;
  for (int c = 0; c < C; ++c) { float v = hp[c]; ss += v * as[c]; sd += v * ad[c]; }
  os[i] = ss; od[i] = sd;
}

static __device__ __forceinline__ void edge_sd(const int* __restrict__ ei, int e,
                                               int& s, int& d) {
  if (e < NEDGES) { s = ei[e]; d = ei[NEDGES + e]; }
  else            { s = d = e - NEDGES; }   // self loops appended
}

// ---------- pass A: segment max ----------
__global__ void k_edge_max(const int* __restrict__ ei,
                           const float* __restrict__ asrc,
                           const float* __restrict__ adst,
                           unsigned int* __restrict__ mEnc, int H) {
  int i = blockIdx.x * blockDim.x + threadIdx.x;
  if (i >= ETOT * H) return;
  int e = i / H, hh = i % H, s, d;
  edge_sd(ei, e, s, d);
  float v = asrc[s * H + hh] + adst[d * H + hh];
  v = (v > 0.f) ? v : NEG_SLOPE * v;
  atomicMax(&mEnc[d * H + hh], enc_f32(v));
}

// ---------- pass B: ex = exp(e - m[dst]); denom[dst] += ex ----------
__global__ void k_edge_expsum(const int* __restrict__ ei,
                              const float* __restrict__ asrc,
                              const float* __restrict__ adst,
                              const unsigned int* __restrict__ mEnc,
                              float* __restrict__ den,
                              float* __restrict__ ex, int H) {
  int i = blockIdx.x * blockDim.x + threadIdx.x;
  if (i >= ETOT * H) return;
  int e = i / H, hh = i % H, s, d;
  edge_sd(ei, e, s, d);
  float v = asrc[s * H + hh] + adst[d * H + hh];
  v = (v > 0.f) ? v : NEG_SLOPE * v;
  float ev = expf(v - dec_f32(mEnc[d * H + hh]));
  ex[i] = ev;
  atomicAdd(&den[d * H + hh], ev);
}

// ---------- pass B2: alpha = ex / den[dst] (once per edge-head) ----------
__global__ void k_norm_alpha(const int* __restrict__ ei,
                             const float* __restrict__ den,
                             float* __restrict__ ex, int H) {
  int i = blockIdx.x * blockDim.x + threadIdx.x;
  if (i >= ETOT * H) return;
  int e = i / H, hh = i % H, s, d;
  edge_sd(ei, e, s, d);
  ex[i] = ex[i] / den[d * H + hh];
}

// ---------- pass C: agg[dst] += h[src] * alpha (float4 per thread) ----------
__global__ void k_edge_agg(const int* __restrict__ ei,
                           const float* __restrict__ hfeat,
                           const float* __restrict__ alpha,
                           float* __restrict__ agg, int H, int C4) {
  int i = blockIdx.x * blockDim.x + threadIdx.x;
  int total = ETOT * H * C4;
  if (i >= total) return;
  int c4 = i % C4;
  int t  = i / C4;
  int hh = t % H;
  int e  = t / H;
  int s, d;
  edge_sd(ei, e, s, d);
  float al = alpha[e * H + hh];
  size_t HC = (size_t)H * C4 * 4;
  const f32x4 hv = *(const f32x4*)(hfeat + (size_t)s * HC + (size_t)hh * C4 * 4 + c4 * 4);
  float* ap = agg + (size_t)d * HC + (size_t)hh * C4 * 4 + c4 * 4;
  atomicAdd(ap + 0, hv[0] * al);
  atomicAdd(ap + 1, hv[1] * al);
  atomicAdd(ap + 2, hv[2] * al);
  atomicAdd(ap + 3, hv[3] * al);
}

// =======================================================================
extern "C" void kernel_launch(void* const* d_in, const int* in_sizes, int n_in,
                              void* d_out, int out_size, void* d_ws, size_t ws_size,
                              hipStream_t stream) {
  const float* x      = (const float*)d_in[0];
  const int*   ei     = (const int*)  d_in[1];   // [2, NEDGES]
  const float* W1     = (const float*)d_in[2];   // [512,1024]
  const float* a_src1 = (const float*)d_in[3];   // [8,128]
  const float* a_dst1 = (const float*)d_in[4];
  const float* b1     = (const float*)d_in[5];   // [1024]
  const float* W2     = (const float*)d_in[6];   // [1024,256]
  const float* a_src2 = (const float*)d_in[7];   // [1,256]
  const float* a_dst2 = (const float*)d_in[8];
  const float* b2     = (const float*)d_in[9];   // [256]
  float* out = (float*)d_out;                    // [10000,256]

  // ---- workspace layout (all sizes multiples of 256B) ----
  char* w = (char*)d_ws;
  const size_t SZ_XB   = (size_t)NNODES * INDIM * 2;
  const size_t SZ_W1B  = (size_t)INDIM * F1 * 2;
  const size_t SZ_H1   = (size_t)NNODES * F1 * 4;
  const size_t SZ_NH1  = (size_t)NNODES * H1 * 4;
  const size_t SZ_EX1  = (size_t)ETOT * H1 * 4;
  const size_t SZ_AGG1 = (size_t)NNODES * F1 * 4;
  const size_t SZ_H1B  = (size_t)NNODES * F1 * 2;

  size_t o = 0;
  __bf16* Xb  = (__bf16*)(w + o);        o += SZ_XB;
  __bf16* W1b = (__bf16*)(w + o);        o += SZ_W1B;
  float*  h1  = (float*) (w + o);        o += SZ_H1;
  float*  as1 = (float*) (w + o);        o += SZ_NH1;
  float*  ad1 = (float*) (w + o);        o += SZ_NH1;
  unsigned int* m1 = (unsigned int*)(w + o); o += SZ_NH1;
  float*  dn1 = (float*) (w + o);        o += SZ_NH1;
  char*   ex1r= (w + o);                 o += SZ_EX1;   // reused for layer-2 smalls
  float*  ag1 = (float*) (w + o);        o += SZ_AGG1;
  __bf16* h1b = (__bf16*)(w + o);        o += SZ_H1B;
  __bf16* W2b = (__bf16*)(w + o);        o += (size_t)F1 * F2 * 2;
  // aliases (layer-1 buffers dead before these are written):
  float*  ex1 = (float*)ex1r;
  float*  h2  = (float*)(w + 0);                     // reuses Xb region
  float*  as2 = (float*)(ex1r + 0);
  float*  ad2 = (float*)(ex1r + 40000);
  unsigned int* m2 = (unsigned int*)(ex1r + 80000);
  float*  dn2 = (float*)(ex1r + 120000);
  float*  ex2 = (float*)(ex1r + 160000);
  (void)ws_size; (void)n_in; (void)in_sizes; (void)out_size;

  auto cdiv = [](int a, int b) { return (a + b - 1) / b; };
  int n;

  // ===================== layer 1 =====================
  n = NNODES * INDIM;
  k_cvt_bf16<<<cdiv(n, 256), 256, 0, stream>>>(x, Xb, n);
  n = INDIM * F1;
  k_cvt_bf16<<<cdiv(n, 256), 256, 0, stream>>>(W1, W1b, n);
  k_gemm_bf16<<<dim3(F1 / 128, cdiv(NNODES, 64)), 256, 0, stream>>>(
      Xb, W1b, h1, NNODES, F1, INDIM);

  k_attn_scores<<<cdiv(NNODES * H1, 256), 256, 0, stream>>>(
      h1, a_src1, a_dst1, as1, ad1, NNODES, H1, C1);

  n = NNODES * H1;
  k_fill_u32<<<cdiv(n, 256), 256, 0, stream>>>(m1, 0u, n);
  k_fill_u32<<<cdiv(n, 256), 256, 0, stream>>>((unsigned int*)dn1, 0u, n);
  n = NNODES * F1;
  k_fill_u32<<<cdiv(n, 256), 256, 0, stream>>>((unsigned int*)ag1, 0u, n);

  n = ETOT * H1;
  k_edge_max   <<<cdiv(n, 256), 256, 0, stream>>>(ei, as1, ad1, m1, H1);
  k_edge_expsum<<<cdiv(n, 256), 256, 0, stream>>>(ei, as1, ad1, m1, dn1, ex1, H1);
  k_norm_alpha <<<cdiv(n, 256), 256, 0, stream>>>(ei, dn1, ex1, H1);
  n = ETOT * H1 * (C1 / 4);
  k_edge_agg   <<<cdiv(n, 256), 256, 0, stream>>>(ei, h1, ex1, ag1, H1, C1 / 4);

  n = NNODES * F1;
  k_elu_cvt<<<cdiv(n, 256), 256, 0, stream>>>(ag1, b1, h1b, n, F1 - 1);

  // ===================== layer 2 =====================
  n = F1 * F2;
  k_cvt_bf16<<<cdiv(n, 256), 256, 0, stream>>>(W2, W2b, n);
  k_gemm_bf16<<<dim3(F2 / 128, cdiv(NNODES, 64)), 256, 0, stream>>>(
      h1b, W2b, h2, NNODES, F2, F1);

  k_attn_scores<<<cdiv(NNODES, 256), 256, 0, stream>>>(
      h2, a_src2, a_dst2, as2, ad2, NNODES, 1, F2);

  n = NNODES;
  k_fill_u32<<<cdiv(n, 256), 256, 0, stream>>>(m2, 0u, n);
  k_fill_u32<<<cdiv(n, 256), 256, 0, stream>>>((unsigned int*)dn2, 0u, n);
  n = NNODES * F2;
  k_fill_u32<<<cdiv(n, 256), 256, 0, stream>>>((unsigned int*)out, 0u, n);

  n = ETOT;
  k_edge_max   <<<cdiv(n, 256), 256, 0, stream>>>(ei, as2, ad2, m2, 1);
  k_edge_expsum<<<cdiv(n, 256), 256, 0, stream>>>(ei, as2, ad2, m2, dn2, ex2, 1);
  k_norm_alpha <<<cdiv(n, 256), 256, 0, stream>>>(ei, dn2, ex2, 1);
  n = ETOT * (F2 / 4);
  k_edge_agg   <<<cdiv(n, 256), 256, 0, stream>>>(ei, h2, ex2, out, 1, F2 / 4);

  n = NNODES * F2;
  k_add_bias<<<cdiv(n, 256), 256, 0, stream>>>(out, b2, n, F2 - 1);
}
